// LSTMModel_17540646437690
// MI455X (gfx1250) — matrix operand order repaired and checked by reading, not compiled, not run on hardware
//
#include <hip/hip_runtime.h>
#include <cstdint>
#include <cstddef>

typedef __attribute__((ext_vector_type(16))) _Float16 v16h;
typedef __attribute__((ext_vector_type(8)))  float    v8f;

#define TSEQ   50
#define DIN    7
#define HID    64
#define NG     256          // 4*H gate columns
#define PRED   30
#define TSTART (TSEQ - PRED)
#define MTILE  16           // batch rows per block (WMMA M)
#define NWAVE  4            // one wave per gate group (i,f,g,o)
#define NTHREADS (NWAVE * 32)

// Hardware-transcendental activations: v_exp_f32 + v_rcp_f32 instead of the
// heavyweight __ocml_tanh_f32 polynomial (these sit on the serial T-loop
// critical path, 80 of them per thread per step).
__device__ __forceinline__ float fast_sigmoid(float x) {
  return __builtin_amdgcn_rcpf(1.0f + __expf(-x));
}
__device__ __forceinline__ float fast_tanh(float x) {
  // tanh(x) = 1 - 2/(exp(2x)+1)
  return 1.0f - 2.0f * __builtin_amdgcn_rcpf(1.0f + __expf(2.0f * x));
}

union FragU { v16h v; uint4 u[2]; };

// A operand (16xK f16, row-major in LDS, ld halves per row), K-tile kt (32 wide).
// CDNA5 A layout: lanes 0-15 -> K = kt*32 + {0..7, 16..23}; lanes 16-31 -> +8.
__device__ __forceinline__ v16h load_a_frag(const _Float16* row_base, int ld,
                                            int kt, int lane) {
  int m  = lane & 15;
  int hi = lane >> 4;
  const _Float16* p = row_base + m * ld + kt * 32 + hi * 8;
  FragU r;
  r.u[0] = *(const uint4*)(p);        // e 0..7  -> K = kt*32 + hi*8 + e
  r.u[1] = *(const uint4*)(p + 16);   // e 8..15 -> K = kt*32 + 16 + hi*8 + e
  return r.v;
}

// B operand from fragment-major LDS: frag f, each lane's 16 halves contiguous
// (two ds_load_b128 per fragment, no per-step gather).
__device__ __forceinline__ v16h load_b_frag(const _Float16* fragbase, int f,
                                            int lane) {
  const _Float16* p = fragbase + (((size_t)f << 5) + lane) * 16;
  FragU r;
  r.u[0] = ((const uint4*)p)[0];
  r.u[1] = ((const uint4*)p)[1];
  return r.v;
}

// Pre-swizzle a [256][Ksrc] f32 weight (gates = X @ W^T) into f16 fragment-major
// B layout: frag (kt,nt): lane L holds column n = nt*16 + (L&15),
// element e holds K = kt*32 + (e>=8 ? 16 : 0) + (L>=16 ? 8 : 0) + (e&7).  K padded with zeros.
__device__ void pack_weight(const float* __restrict__ W, int Ksrc, int nkt,
                            _Float16* __restrict__ dst, int tid) {
  int total = nkt * 16 * 512;             // frags * 32 lanes * 16 halves
  for (int idx = tid; idx < total; idx += NTHREADS) {
    int f    = idx >> 9;
    int r    = idx & 511;
    int lane = r >> 4;
    int e    = r & 15;
    int nt   = f & 15;
    int kt   = f >> 4;
    int n    = (nt << 4) + (lane & 15);
    int k    = (kt << 5) + ((e >> 3) << 4) + ((lane >> 4) << 3) + (e & 7);
    float v  = (k < Ksrc) ? W[n * Ksrc + k] : 0.0f;
    dst[idx] = (_Float16)v;
  }
}

__global__ __launch_bounds__(NTHREADS) void lstm2_fused(
    const float* __restrict__ x,
    const float* __restrict__ Wih0, const float* __restrict__ Whh0,
    const float* __restrict__ bih0, const float* __restrict__ bhh0,
    const float* __restrict__ Wih1, const float* __restrict__ Whh1,
    const float* __restrict__ bih1, const float* __restrict__ bhh1,
    const float* __restrict__ Wfc,  const float* __restrict__ bfc,
    float* __restrict__ out) {
  // Weights pre-swizzled to f16 fragment-major (resident all 50 steps).
  __shared__ __align__(16) _Float16 fWih0[16 * 512];   // K=32 (pad of 7)
  __shared__ __align__(16) _Float16 fWhh0[32 * 512];   // K=64
  __shared__ __align__(16) _Float16 fWih1[32 * 512];   // K=64
  __shared__ __align__(16) _Float16 fWhh1[32 * 512];   // K=64
  __shared__ __align__(16) _Float16 xt_s[MTILE * 32];  // x tile, K padded to 32
  __shared__ __align__(16) _Float16 hA_s[MTILE * HID]; // layer-0 hidden (A op)
  __shared__ __align__(16) _Float16 hB_s[MTILE * HID]; // layer-1 hidden (A op)
  __shared__ __align__(16) float    gates_s[MTILE * NG];
  __shared__ float bias0_s[NG], bias1_s[NG];
  __shared__ float wfc_s[2 * HID];
  __shared__ float bfc_s[2];

  const int tid  = threadIdx.x;
  const int lane = tid & 31;
  const int w    = tid >> 5;            // wave id == gate group
  const int b0   = blockIdx.x * MTILE;  // batch-row base of this block

  pack_weight(Wih0, DIN, 1, fWih0, tid);
  pack_weight(Whh0, HID, 2, fWhh0, tid);
  pack_weight(Wih1, HID, 2, fWih1, tid);
  pack_weight(Whh1, HID, 2, fWhh1, tid);
  for (int i = tid; i < NG; i += NTHREADS) {
    bias0_s[i] = bih0[i] + bhh0[i];
    bias1_s[i] = bih1[i] + bhh1[i];
  }
  for (int i = tid; i < 2 * HID; i += NTHREADS) wfc_s[i] = Wfc[i];
  if (tid < 2) bfc_s[tid] = bfc[tid];
  for (int i = tid; i < MTILE * HID; i += NTHREADS) {
    hA_s[i] = (_Float16)0.0f;
    hB_s[i] = (_Float16)0.0f;
  }
  // ---- stage x tile for t = 0 (K padded 7 -> 32 with zeros, f16) ----
  for (int i = tid; i < MTILE * 32; i += NTHREADS) {
    int m = i >> 5, k = i & 31;
    xt_s[i] = (_Float16)((k < DIN)
                  ? x[(size_t)(b0 + m) * TSEQ * DIN + k] : 0.0f);
  }
  __syncthreads();

  // Per-thread cell state: row m = tid&15, cols j = (tid>>4)*8 .. +7
  const int um = tid & 15;
  const int uj = (tid >> 4) << 3;
  float c0[8], c1[8];
#pragma unroll
  for (int q = 0; q < 8; ++q) { c0[q] = 0.0f; c1[q] = 0.0f; }

  const int gcol0 = w * 64;       // this wave's first gate column
  const int nfrb  = w * 4;        // first N-fragment index

#pragma unroll 1
  for (int t = 0; t < TSEQ; ++t) {
    v8f acc[4];

    // ================= layer 0: gates = x@Wih0^T + hA@Whh0^T + b =========
#pragma unroll
    for (int nt = 0; nt < 4; ++nt) {
      float bb = bias0_s[gcol0 + nt * 16 + (lane & 15)];
      v8f a = {bb, bb, bb, bb, bb, bb, bb, bb};
      acc[nt] = a;
    }
    {
      v16h ax  = load_a_frag(xt_s, 32, 0, lane);
      v16h ah0 = load_a_frag(hA_s, HID, 0, lane);
      v16h ah1 = load_a_frag(hA_s, HID, 1, lane);
#pragma unroll
      for (int nt = 0; nt < 4; ++nt) {
        int nf = nfrb + nt;
        acc[nt] = __builtin_amdgcn_wmma_f32_16x16x32_f16(
            false, ax, false, load_b_frag(fWih0, nf, lane), (short)0, acc[nt],
            false, false);
        acc[nt] = __builtin_amdgcn_wmma_f32_16x16x32_f16(
            false, ah0, false, load_b_frag(fWhh0, nf, lane), (short)0, acc[nt],
            false, false);
        acc[nt] = __builtin_amdgcn_wmma_f32_16x16x32_f16(
            false, ah1, false, load_b_frag(fWhh0, 16 + nf, lane), (short)0,
            acc[nt], false, false);
      }
    }
    // C/D layout: VGPR r -> row r + 8*(lane>=16), col = lane&15
#pragma unroll
    for (int nt = 0; nt < 4; ++nt) {
      int n = gcol0 + nt * 16 + (lane & 15);
      int rbase = (lane >> 4) * 8;
#pragma unroll
      for (int r = 0; r < 8; ++r) gates_s[(rbase + r) * NG + n] = acc[nt][r];
    }
    __syncthreads();

    // ---- pipelined stage of x[t+1]: xt_s is dead now; the global-load
    // latency hides behind the layer-0 update and the whole layer-1 GEMM.
    // Next read of xt_s is after the upd-1 barrier, so ordering holds. ----
    if (t + 1 < TSEQ) {
      for (int i = tid; i < MTILE * 32; i += NTHREADS) {
        int m = i >> 5, k = i & 31;
        xt_s[i] = (_Float16)((k < DIN)
                      ? x[(size_t)(b0 + m) * TSEQ * DIN + (t + 1) * DIN + k]
                      : 0.0f);
      }
      if (t + 2 < TSEQ)   // warm L2/WGP$ one more step ahead
        __builtin_prefetch(&x[(size_t)(b0 + (tid & 15)) * TSEQ * DIN +
                              (t + 2) * DIN], 0, 0);
    }

    // ---- layer-0 elementwise update: c0, hA ----
#pragma unroll
    for (int q = 0; q < 8; ++q) {
      int j = uj + q;
      float gi = fast_sigmoid(gates_s[um * NG + j]);
      float gf = fast_sigmoid(gates_s[um * NG + 64 + j]);
      float gg = fast_tanh(gates_s[um * NG + 128 + j]);
      float go = fast_sigmoid(gates_s[um * NG + 192 + j]);
      float c  = gf * c0[q] + gi * gg;
      c0[q]    = c;
      hA_s[um * HID + j] = (_Float16)(go * fast_tanh(c));
    }
    __syncthreads();

    // ================ layer 1: gates = hA@Wih1^T + hB@Whh1^T + b =========
#pragma unroll
    for (int nt = 0; nt < 4; ++nt) {
      float bb = bias1_s[gcol0 + nt * 16 + (lane & 15)];
      v8f a = {bb, bb, bb, bb, bb, bb, bb, bb};
      acc[nt] = a;
    }
    {
      v16h aA0 = load_a_frag(hA_s, HID, 0, lane);
      v16h aA1 = load_a_frag(hA_s, HID, 1, lane);
      v16h aB0 = load_a_frag(hB_s, HID, 0, lane);
      v16h aB1 = load_a_frag(hB_s, HID, 1, lane);
#pragma unroll
      for (int nt = 0; nt < 4; ++nt) {
        int nf = nfrb + nt;
        acc[nt] = __builtin_amdgcn_wmma_f32_16x16x32_f16(
            false, aA0, false, load_b_frag(fWih1, nf, lane), (short)0, acc[nt],
            false, false);
        acc[nt] = __builtin_amdgcn_wmma_f32_16x16x32_f16(
            false, aA1, false, load_b_frag(fWih1, 16 + nf, lane), (short)0,
            acc[nt], false, false);
        acc[nt] = __builtin_amdgcn_wmma_f32_16x16x32_f16(
            false, aB0, false, load_b_frag(fWhh1, nf, lane), (short)0, acc[nt],
            false, false);
        acc[nt] = __builtin_amdgcn_wmma_f32_16x16x32_f16(
            false, aB1, false, load_b_frag(fWhh1, 16 + nf, lane), (short)0,
            acc[nt], false, false);
      }
    }
#pragma unroll
    for (int nt = 0; nt < 4; ++nt) {
      int n = gcol0 + nt * 16 + (lane & 15);
      int rbase = (lane >> 4) * 8;
#pragma unroll
      for (int r = 0; r < 8; ++r) gates_s[(rbase + r) * NG + n] = acc[nt][r];
    }
    __syncthreads();

    // ---- layer-1 elementwise update: c1, hB ----
#pragma unroll
    for (int q = 0; q < 8; ++q) {
      int j = uj + q;
      float gi = fast_sigmoid(gates_s[um * NG + j]);
      float gf = fast_sigmoid(gates_s[um * NG + 64 + j]);
      float gg = fast_tanh(gates_s[um * NG + 128 + j]);
      float go = fast_sigmoid(gates_s[um * NG + 192 + j]);
      float c  = gf * c1[q] + gi * gg;
      c1[q]    = c;
      hB_s[um * HID + j] = (_Float16)(go * fast_tanh(c));
    }
    __syncthreads();

    // ---- FC head on last PRED steps: out[b, t-20, :] = hB @ Wfc^T + bfc ----
    if (t >= TSTART && tid < 32) {
      int m = tid & 15, o = tid >> 4;
      float s = bfc_s[o];
      for (int j = 0; j < HID; ++j)
        s += (float)hB_s[m * HID + j] * wfc_s[o * HID + j];
      out[(((size_t)(b0 + m)) * PRED + (t - TSTART)) * 2 + o] = s;
    }
    // hB_s is not rewritten until after several barriers next iteration.
  }
}

extern "C" void kernel_launch(void* const* d_in, const int* in_sizes, int n_in,
                              void* d_out, int out_size, void* d_ws,
                              size_t ws_size, hipStream_t stream) {
  const float* x    = (const float*)d_in[0];
  const float* Wih0 = (const float*)d_in[1];
  const float* Whh0 = (const float*)d_in[2];
  const float* bih0 = (const float*)d_in[3];
  const float* bhh0 = (const float*)d_in[4];
  const float* Wih1 = (const float*)d_in[5];
  const float* Whh1 = (const float*)d_in[6];
  const float* bih1 = (const float*)d_in[7];
  const float* bhh1 = (const float*)d_in[8];
  const float* Wfc  = (const float*)d_in[9];
  const float* bfc  = (const float*)d_in[10];
  float* out = (float*)d_out;

  int B = in_sizes[0] / (TSEQ * DIN);   // 8192
  dim3 grid(B / MTILE);                 // 512 independent batch tiles
  lstm2_fused<<<grid, NTHREADS, 0, stream>>>(x, Wih0, Whh0, bih0, bhh0, Wih1,
                                             Whh1, bih1, bhh1, Wfc, bfc, out);
}
